// TrueLoRAAttentionExpertGPU_858993459664
// MI455X (gfx1250) — compile-verified, matrix operand order checked by reading, not run on hardware
//
#include <hip/hip_runtime.h>
#include <hip/hip_bf16.h>

// ---------------- problem constants ----------------
#define BATCH 2
#define SEQ   2048
#define HDIM  1024
#define NHEAD 16
#define HEADD 64
#define RANK  8
#define MTOT  (BATCH * SEQ)          // 4096 rows
#define LORA_SCALE_F 8.0f            // 64/8
#define INV_SCALE 0.125f             // 1/sqrt(64)

typedef __attribute__((ext_vector_type(16))) __bf16 v16bf;
typedef __attribute__((ext_vector_type(8)))  __bf16 v8bf;
typedef __attribute__((ext_vector_type(8)))  float  v8f;

__device__ __forceinline__ unsigned short f2bf(float f) {
    union { float f; unsigned int u; } v; v.f = f;
    unsigned int r = v.u + 0x7FFFu + ((v.u >> 16) & 1u);   // RNE
    return (unsigned short)(r >> 16);
}
__device__ __forceinline__ float bf2f(unsigned short h) {
    union { unsigned int u; float f; } v; v.u = ((unsigned int)h) << 16;
    return v.f;
}
__device__ __forceinline__ v16bf mk16(v8bf lo, v8bf hi) {
    v16bf r;
#pragma unroll
    for (int i = 0; i < 8; ++i) { r[i] = lo[i]; r[i + 8] = hi[i]; }
    return r;
}
// 16x32 A / 32x16 B bf16 fragment: lane l16 owns row/col, half-wave splits K
// into {0..7,16..23} (lanes 0-15) and {8..15,24..31} (lanes 16-31).
__device__ __forceinline__ v16bf load_frag(const __bf16* base) {
    v8bf lo = *(const v8bf*)(base);
    v8bf hi = *(const v8bf*)(base + 16);
    return mk16(lo, hi);
}

// ---------------- conversion kernels ----------------
__global__ void k_cvt_bf16(const float* __restrict__ x,
                           unsigned short* __restrict__ y, int n) {
    int i = blockIdx.x * blockDim.x + threadIdx.x;
    if (i < n) y[i] = f2bf(x[i]);
}

// W [K rows, N cols] fp32 -> Wt [N rows, K cols] bf16 (B-operand contiguous)
__global__ void k_cvt_transpose(const float* __restrict__ W,
                                unsigned short* __restrict__ Wt, int K, int N) {
    int i = blockIdx.x * blockDim.x + threadIdx.x;
    if (i < K * N) {
        int k = i / N, n = i - k * N;
        Wt[n * K + k] = f2bf(W[i]);
    }
}

// vbuf [B*S, H] bf16 -> vT [B][NH][HEADD][SEQ] bf16 (ctx B-operand contiguous)
__global__ void k_transpose_v(const unsigned short* __restrict__ vbuf,
                              unsigned short* __restrict__ vT) {
    int i = blockIdx.x * blockDim.x + threadIdx.x;   // over B*S*H
    if (i >= MTOT * HDIM) return;
    int s  = i / HDIM;                 // 0..MTOT-1 (b*SEQ + seq)
    int c  = i - s * HDIM;             // 0..HDIM-1 (h*HEADD + d)
    int b  = s / SEQ;
    int sq = s - b * SEQ;
    int h  = c >> 6;                   // /HEADD
    int d  = c & 63;
    vT[(((size_t)(b * NHEAD + h) * HEADD + d) * SEQ) + sq] = vbuf[i];
}

// ---------------- LoRA down-projection: T[m,r] = sum_k X[m,k]*A[k,r] ----------------
__global__ void k_loraT_f32(const float* __restrict__ X, const float* __restrict__ A,
                            float* __restrict__ T, int M, int K) {
    int i = blockIdx.x * blockDim.x + threadIdx.x;
    if (i >= M * RANK) return;
    int m = i >> 3, r = i & 7;
    float s = 0.f;
    for (int k = 0; k < K; ++k) s += X[m * K + k] * A[k * RANK + r];
    T[i] = s;
}
__global__ void k_loraT_bf16(const unsigned short* __restrict__ X,
                             const float* __restrict__ A,
                             float* __restrict__ T, int M, int K) {
    int i = blockIdx.x * blockDim.x + threadIdx.x;
    if (i >= M * RANK) return;
    int m = i >> 3, r = i & 7;
    float s = 0.f;
    for (int k = 0; k < K; ++k) s += bf2f(X[m * K + k]) * A[k * RANK + r];
    T[i] = s;
}

// ---------------- WMMA GEMM + fused rank-8 LoRA epilogue ----------------
// Y[m,n] = sum_k X[m,k]*Wt[n,k]  +  LORA_SCALE * sum_r T[m,r]*Bl[r,n]
// One wave computes a 32(M) x 64(N) tile: 2 A frags, 4 B frags, 8 WMMAs / K-step.
// Grid = (M/32)*(N/64) waves, 8 waves/block.
__global__ void k_gemm_lora(const unsigned short* __restrict__ Xb,
                            const unsigned short* __restrict__ Wtb,
                            const float* __restrict__ T,
                            const float* __restrict__ Bl,
                            void* __restrict__ Yout, int outIsBf16,
                            int M, int N, int K) {
    const __bf16* X  = (const __bf16*)Xb;
    const __bf16* Wt = (const __bf16*)Wtb;

    int wave  = blockIdx.x * 8 + (threadIdx.x >> 5);
    int lane  = threadIdx.x & 31;
    int hf    = lane >> 4;          // half-wave
    int l16   = lane & 15;
    int nT64  = N / 64;
    int mTile = wave / nT64;
    int nTile = wave - mTile * nT64;
    int row0  = mTile * 32 + l16;   // A rows for this lane (two 16-row subtiles)
    int row1  = row0 + 16;

    v8f acc[2][4];
#pragma unroll
    for (int s = 0; s < 2; ++s)
#pragma unroll
        for (int f = 0; f < 4; ++f) acc[s][f] = (v8f){};

#pragma unroll 2
    for (int kb = 0; kb < K; kb += 32) {
        int kOff = kb + hf * 8;
        v16bf A0 = load_frag(X + (size_t)row0 * K + kOff);
        v16bf A1 = load_frag(X + (size_t)row1 * K + kOff);

        int col0 = nTile * 64 + l16;
#pragma unroll
        for (int f = 0; f < 4; ++f) {
            v16bf B = load_frag(Wt + (size_t)(col0 + f * 16) * K + kOff);
            acc[0][f] = __builtin_amdgcn_wmma_f32_16x16x32_bf16(
                false, A0, false, B, (short)0, acc[0][f], false, false);
            acc[1][f] = __builtin_amdgcn_wmma_f32_16x16x32_bf16(
                false, A1, false, B, (short)0, acc[1][f], false, false);
        }
    }

    // epilogue: rank-8 LoRA add + store (C/D layout: VGPR r -> row r + hf*8)
#pragma unroll
    for (int s = 0; s < 2; ++s) {
#pragma unroll
        for (int r = 0; r < 8; ++r) {
            int m = mTile * 32 + s * 16 + r + hf * 8;
            float t[RANK];
#pragma unroll
            for (int j = 0; j < RANK; ++j) t[j] = T[m * RANK + j];
#pragma unroll
            for (int f = 0; f < 4; ++f) {
                int n = nTile * 64 + f * 16 + l16;
                float lo = 0.f;
#pragma unroll
                for (int j = 0; j < RANK; ++j) lo += t[j] * Bl[j * N + n];
                float y = acc[s][f][r] + LORA_SCALE_F * lo;
                if (outIsBf16) ((unsigned short*)Yout)[(size_t)m * N + n] = f2bf(y);
                else           ((float*)Yout)[(size_t)m * N + n] = y;
            }
        }
    }
}

// ---------------- flash attention ----------------
// One wave per (b, head, 16-row q tile); online softmax; k-tile of 32.
// scores: A=q(16x32 over head-dim), B=k^T (contiguous rows of K).
// ctx:    A=P(16x32 via per-wave LDS), B=V^T from pre-transposed global vT.
__global__ void k_attn(const unsigned short* __restrict__ qb,
                       const unsigned short* __restrict__ kb_,
                       const unsigned short* __restrict__ vTb,
                       const float* __restrict__ maskp,
                       unsigned short* __restrict__ ctxb) {
    __shared__ __align__(16) unsigned short ldsP[8][16 * 32];   // per-wave P tile

    int wv    = threadIdx.x >> 5;
    int lane  = threadIdx.x & 31;
    int hf    = lane >> 4;
    int l16   = lane & 15;

    int wave  = blockIdx.x * 8 + wv;
    int qTile = wave & 127;               // S/16 = 128
    int h     = (wave >> 7) & 15;
    int b     = wave >> 11;

    const __bf16* Q  = (const __bf16*)qb;
    const __bf16* K  = (const __bf16*)kb_;
    const __bf16* VT = (const __bf16*)vTb;    // [B][NH][HEADD][SEQ]
    const __bf16* P  = (const __bf16*)&ldsP[wv][0];

    // persistent q A-fragments (head-dim split 0..31 / 32..63)
    size_t qrow = ((size_t)(b * SEQ + qTile * 16 + l16)) * HDIM + h * HEADD;
    v16bf Aq0 = load_frag(Q + qrow + 0  + hf * 8);
    v16bf Aq1 = load_frag(Q + qrow + 32 + hf * 8);

    // per-dim-column V^T bases: column d = f*16 + l16
    size_t vtHead = (size_t)(b * NHEAD + h) * HEADD * SEQ;

    float mrun[8], lrun[8];
    v8f c0 = {}, c1 = {}, c2 = {}, c3 = {};
#pragma unroll
    for (int r = 0; r < 8; ++r) { mrun[r] = -1e30f; lrun[r] = 0.f; }

#pragma unroll 1
    for (int kb = 0; kb < SEQ; kb += 32) {
        // ---- scores: two 16x16 column tiles ----
        v8f s0 = {}, s1 = {};
        {
            size_t krow0 = ((size_t)(b * SEQ + kb + l16)) * HDIM + h * HEADD;
            size_t krow1 = ((size_t)(b * SEQ + kb + 16 + l16)) * HDIM + h * HEADD;
            v16bf Bk;
            Bk = load_frag(K + krow0 + 0  + hf * 8);
            s0 = __builtin_amdgcn_wmma_f32_16x16x32_bf16(false, Aq0, false, Bk, (short)0, s0, false, false);
            Bk = load_frag(K + krow0 + 32 + hf * 8);
            s0 = __builtin_amdgcn_wmma_f32_16x16x32_bf16(false, Aq1, false, Bk, (short)0, s0, false, false);
            Bk = load_frag(K + krow1 + 0  + hf * 8);
            s1 = __builtin_amdgcn_wmma_f32_16x16x32_bf16(false, Aq0, false, Bk, (short)0, s1, false, false);
            Bk = load_frag(K + krow1 + 32 + hf * 8);
            s1 = __builtin_amdgcn_wmma_f32_16x16x32_bf16(false, Aq1, false, Bk, (short)0, s1, false, false);
        }
        // scale + additive mask (per k column = per lane)
        float madd0 = (1.0f - maskp[b * SEQ + kb + l16]) * -10000.0f;
        float madd1 = (1.0f - maskp[b * SEQ + kb + 16 + l16]) * -10000.0f;
#pragma unroll
        for (int r = 0; r < 8; ++r) {
            s0[r] = s0[r] * INV_SCALE + madd0;
            s1[r] = s1[r] * INV_SCALE + madd1;
        }
        // ---- online softmax (row stats live per half-wave, replicated) ----
#pragma unroll
        for (int r = 0; r < 8; ++r) {
            float mx = fmaxf(s0[r], s1[r]);
            mx = fmaxf(mx, __shfl_xor(mx, 1));
            mx = fmaxf(mx, __shfl_xor(mx, 2));
            mx = fmaxf(mx, __shfl_xor(mx, 4));
            mx = fmaxf(mx, __shfl_xor(mx, 8));
            float mnew  = fmaxf(mrun[r], mx);
            float alpha = __expf(mrun[r] - mnew);
            float p0 = __expf(s0[r] - mnew);
            float p1 = __expf(s1[r] - mnew);
            float rs = p0 + p1;
            rs += __shfl_xor(rs, 1);
            rs += __shfl_xor(rs, 2);
            rs += __shfl_xor(rs, 4);
            rs += __shfl_xor(rs, 8);
            lrun[r] = lrun[r] * alpha + rs;
            mrun[r] = mnew;
            c0[r] *= alpha; c1[r] *= alpha; c2[r] *= alpha; c3[r] *= alpha;
            // stash P (bf16) into LDS, row-major 16x32 (C-layout -> A-layout bridge)
            ldsP[wv][(r + hf * 8) * 32 + l16]      = f2bf(p0);
            ldsP[wv][(r + hf * 8) * 32 + 16 + l16] = f2bf(p1);
        }
        // ---- ctx += P @ V : A from LDS P, B from pre-transposed global V^T ----
        v16bf Ap = load_frag(P + (size_t)l16 * 32 + hf * 8);
        size_t vk = kb + hf * 8;
        v16bf Bv;
        Bv = load_frag(VT + vtHead + (size_t)(l16     ) * SEQ + vk);
        c0 = __builtin_amdgcn_wmma_f32_16x16x32_bf16(false, Ap, false, Bv, (short)0, c0, false, false);
        Bv = load_frag(VT + vtHead + (size_t)(l16 + 16) * SEQ + vk);
        c1 = __builtin_amdgcn_wmma_f32_16x16x32_bf16(false, Ap, false, Bv, (short)0, c1, false, false);
        Bv = load_frag(VT + vtHead + (size_t)(l16 + 32) * SEQ + vk);
        c2 = __builtin_amdgcn_wmma_f32_16x16x32_bf16(false, Ap, false, Bv, (short)0, c2, false, false);
        Bv = load_frag(VT + vtHead + (size_t)(l16 + 48) * SEQ + vk);
        c3 = __builtin_amdgcn_wmma_f32_16x16x32_bf16(false, Ap, false, Bv, (short)0, c3, false, false);
    }

    // ---- normalize and store ctx (bf16, [B,S,H] with heads concatenated) ----
#pragma unroll
    for (int r = 0; r < 8; ++r) {
        float inv = 1.0f / lrun[r];
        size_t orow = ((size_t)(b * SEQ + qTile * 16 + r + hf * 8)) * HDIM + h * HEADD;
        ctxb[orow + 0  + l16] = f2bf(c0[r] * inv);
        ctxb[orow + 16 + l16] = f2bf(c1[r] * inv);
        ctxb[orow + 32 + l16] = f2bf(c2[r] * inv);
        ctxb[orow + 48 + l16] = f2bf(c3[r] * inv);
    }
}

// ---------------- host side ----------------
extern "C" void kernel_launch(void* const* d_in, const int* in_sizes, int n_in,
                              void* d_out, int out_size, void* d_ws, size_t ws_size,
                              hipStream_t stream) {
    const float* hs   = (const float*)d_in[0];
    const float* mask = (const float*)d_in[1];
    const float* Wq   = (const float*)d_in[2];
    const float* Wk   = (const float*)d_in[3];
    const float* Wv   = (const float*)d_in[4];
    const float* Wo   = (const float*)d_in[5];
    const float* Aq   = (const float*)d_in[6];
    const float* Bq   = (const float*)d_in[7];
    const float* Ak   = (const float*)d_in[8];
    const float* Bk   = (const float*)d_in[9];
    const float* Av   = (const float*)d_in[10];
    const float* Bv   = (const float*)d_in[11];
    const float* Ao   = (const float*)d_in[12];
    const float* Bo   = (const float*)d_in[13];

    char* ws = (char*)d_ws;
    size_t off = 0;
    auto alloc = [&](size_t bytes) { char* p = ws + off; off += (bytes + 255) & ~size_t(255); return p; };

    unsigned short* hsb  = (unsigned short*)alloc((size_t)MTOT * HDIM * 2);
    unsigned short* wqt  = (unsigned short*)alloc((size_t)HDIM * HDIM * 2);
    unsigned short* wkt  = (unsigned short*)alloc((size_t)HDIM * HDIM * 2);
    unsigned short* wvt  = (unsigned short*)alloc((size_t)HDIM * HDIM * 2);
    unsigned short* wot  = (unsigned short*)alloc((size_t)HDIM * HDIM * 2);
    unsigned short* qb   = (unsigned short*)alloc((size_t)MTOT * HDIM * 2);
    unsigned short* kbuf = (unsigned short*)alloc((size_t)MTOT * HDIM * 2);
    unsigned short* vbuf = (unsigned short*)alloc((size_t)MTOT * HDIM * 2);
    unsigned short* vT   = (unsigned short*)alloc((size_t)MTOT * HDIM * 2);
    unsigned short* ctxb = (unsigned short*)alloc((size_t)MTOT * HDIM * 2);
    float*          Tl   = (float*)alloc((size_t)MTOT * RANK * 4);

    const int nHS = MTOT * HDIM;       // 4,194,304
    const int nW  = HDIM * HDIM;       // 1,048,576

    k_cvt_bf16<<<(nHS + 255) / 256, 256, 0, stream>>>(hs, hsb, nHS);
    k_cvt_transpose<<<(nW + 255) / 256, 256, 0, stream>>>(Wq, wqt, HDIM, HDIM);
    k_cvt_transpose<<<(nW + 255) / 256, 256, 0, stream>>>(Wk, wkt, HDIM, HDIM);
    k_cvt_transpose<<<(nW + 255) / 256, 256, 0, stream>>>(Wv, wvt, HDIM, HDIM);
    k_cvt_transpose<<<(nW + 255) / 256, 256, 0, stream>>>(Wo, wot, HDIM, HDIM);

    const int gemmBlocks = (MTOT / 32) * (HDIM / 64) / 8;   // 256
    const int loraBlocks = (MTOT * RANK + 255) / 256;       // 128

    // q
    k_loraT_f32<<<loraBlocks, 256, 0, stream>>>(hs, Aq, Tl, MTOT, HDIM);
    k_gemm_lora<<<gemmBlocks, 256, 0, stream>>>(hsb, wqt, Tl, Bq, qb, 1, MTOT, HDIM, HDIM);
    // k
    k_loraT_f32<<<loraBlocks, 256, 0, stream>>>(hs, Ak, Tl, MTOT, HDIM);
    k_gemm_lora<<<gemmBlocks, 256, 0, stream>>>(hsb, wkt, Tl, Bk, kbuf, 1, MTOT, HDIM, HDIM);
    // v
    k_loraT_f32<<<loraBlocks, 256, 0, stream>>>(hs, Av, Tl, MTOT, HDIM);
    k_gemm_lora<<<gemmBlocks, 256, 0, stream>>>(hsb, wvt, Tl, Bv, vbuf, 1, MTOT, HDIM, HDIM);

    // V pre-transpose for the ctx WMMA B-operand
    k_transpose_v<<<(nHS + 255) / 256, 256, 0, stream>>>(vbuf, vT);

    // attention: B*NH*(S/16) = 4096 waves -> 512 blocks of 8 waves
    k_attn<<<512, 256, 0, stream>>>(qb, kbuf, vT, mask, ctxb);

    // output projection
    k_loraT_bf16<<<loraBlocks, 256, 0, stream>>>(ctxb, Ao, Tl, MTOT, HDIM);
    k_gemm_lora<<<gemmBlocks, 256, 0, stream>>>(ctxb, wot, Tl, Bo, d_out, 0, MTOT, HDIM, HDIM);
}
